// BlockAxialDown_9560597201241
// MI455X (gfx1250) — matrix operand order repaired and checked by reading, not compiled, not run on hardware
//
#include <hip/hip_runtime.h>

typedef __attribute__((ext_vector_type(16))) _Float16 v16h;
typedef __attribute__((ext_vector_type(8)))  float    v8f;

// ---------------- WMMA micro-kernel helpers (CDNA5 wave32 layouts) ----------------
__device__ __forceinline__ v8f wmma_f16(v16h a, v16h b, v8f c) {
  return __builtin_amdgcn_wmma_f32_16x16x32_f16(false, a, false, b, (short)0, c, false, false);
}

// A fragment: 16x32 f16, row-major in LDS (lda elems). lanes 0-15: M rows, K 0..7 & 16..23;
// lanes 16-31: same rows, K 8..15 & 24..31.
__device__ __forceinline__ v16h ldA(const _Float16* A, int lda, int m0, int ko, int lane) {
  const _Float16* p = A + (m0 + (lane & 15)) * lda + ko;
  int kb = (lane & 16) ? 8 : 0;
  v16h a;
#pragma unroll
  for (int h = 0; h < 8; ++h) { a[h] = p[kb + h]; a[8 + h] = p[16 + kb + h]; }
  return a;
}

// Same as ldA but source is f32 in LDS (converted per element).
__device__ __forceinline__ v16h ldA_f32(const float* A, int lda, int m0, int ko, int lane) {
  const float* p = A + (m0 + (lane & 15)) * lda + ko;
  int kb = (lane & 16) ? 8 : 0;
  v16h a;
#pragma unroll
  for (int h = 0; h < 8; ++h) { a[h] = (_Float16)p[kb + h]; a[8 + h] = (_Float16)p[16 + kb + h]; }
  return a;
}

// B fragment: 32x16 f16, stored TRANSPOSED in LDS (Bt[n*ldb + k], contiguous in k).
// lanes 0-15: N cols, K 0..7 & 16..23; lanes 16-31: same cols, K 8..15 & 24..31.
__device__ __forceinline__ v16h ldBt(const _Float16* Bt, int ldb, int ko, int n0, int lane) {
  const _Float16* p = Bt + (n0 + (lane & 15)) * ldb + ko;
  int kb = (lane & 16) ? 8 : 0;
  v16h b;
#pragma unroll
  for (int h = 0; h < 8; ++h) { b[h] = p[kb + h]; b[8 + h] = p[16 + kb + h]; }
  return b;
}

// D/C layout: lanes 0-15 N cols with M=r; lanes 16-31 same cols with M=r+8.
__device__ __forceinline__ void stD_f16(_Float16* D, int ldd, int m0, int n0, const v8f& acc, int lane) {
  _Float16* p = D + (m0 + ((lane & 16) ? 8 : 0)) * ldd + n0 + (lane & 15);
#pragma unroll
  for (int r = 0; r < 8; ++r) p[r * ldd] = (_Float16)acc[r];
}
__device__ __forceinline__ void stD_f16_T(_Float16* Dt, int ldd, int m0, int n0, const v8f& acc, int lane) {
  _Float16* p = Dt + (n0 + (lane & 15)) * ldd + m0 + ((lane & 16) ? 8 : 0);
#pragma unroll
  for (int r = 0; r < 8; ++r) p[r] = (_Float16)acc[r];
}
__device__ __forceinline__ void stD_f32s(float* D, int ldd, int m0, int n0, const v8f& acc, float s, int lane) {
  float* p = D + (m0 + ((lane & 16) ? 8 : 0)) * ldd + n0 + (lane & 15);
#pragma unroll
  for (int r = 0; r < 8; ++r) p[r * ldd] = acc[r] * s;
}

// Padded LDS strides (row stride % 64 banks == 4 -> fragment row-lanes fan across banks)
#define LDH  136   // f16 panels, stride 272 B
#define LDD  132   // f32 dots panel, stride 528 B
#define LDC  264   // f16 conv panels, stride 528 B

// ---------------- Kernel 1: maxpool2x2 + NCHW -> NHWC ----------------
__global__ __launch_bounds__(256) void pool_kernel(const float* __restrict__ x,
                                                   float* __restrict__ xc) {
  __shared__ float tile[32][33];
  int w20 = blockIdx.x * 32;
  int h2  = blockIdx.y;
  int b   = blockIdx.z >> 2;
  int c0  = (blockIdx.z & 3) * 32;
  int tid = threadIdx.x;
  int w2  = tid & 31;
  int cl0 = tid >> 5;  // 0..7
#pragma unroll
  for (int pass = 0; pass < 4; ++pass) {
    int cl = cl0 + pass * 8;
    const float* r0 = x + (((b * 128 + c0 + cl) * 256 + 2 * h2) * 256) + 2 * (w20 + w2);
    tile[cl][w2] = fmaxf(fmaxf(r0[0], r0[1]), fmaxf(r0[256], r0[257]));
  }
  __syncthreads();
  for (int i = tid; i < 1024; i += 256) {
    int w2o = i >> 5, c = i & 31;
    xc[(((b * 128 + h2) * 128) + w20 + w2o) * 128 + c0 + c] = tile[c][w2o];
  }
}

// ---------------- Kernel 2: axial self-attention, both directions in one launch ----------------
// blockIdx.y==0: attend along H (s = w2) with *_h weights -> ah
// blockIdx.y==1: attend along W (s = h2) with *_w weights -> aw
__global__ __launch_bounds__(256) void attn_kernel(
    const float* __restrict__ xc,
    const float* __restrict__ Wq_h, const float* __restrict__ Wkv_h,
    const float* __restrict__ Wout_h, const float* __restrict__ bout_h,
    const float* __restrict__ Wq_w, const float* __restrict__ Wkv_w,
    const float* __restrict__ Wout_w, const float* __restrict__ bout_w,
    float* __restrict__ ah, float* __restrict__ aw)
{
  // Region overlays (bytes):
  //   [0      , 34816 ) Xs   (f16 128xLDH)      -- phase A   \__ DOTS (f32 128xLDD = 67584 B)
  //   [34816  , 69632 ) Wqt  (f16 128xLDH)      -- phase A   /   in phase B
  //   [69632  , 139264) Wkvt (f16 256xLDH)      -- phase A; Os (34816) @69632, Woutt (34816) @104448
  //   [139264 , 174080) Qs ; [174080,208896) Ks ; [208896,243712) Vts (transposed)
  extern __shared__ char smem[];
  _Float16* Xs    = (_Float16*)(smem);
  float*    DOTS  = (float*)   (smem);
  _Float16* Wqt   = (_Float16*)(smem + 34816);
  _Float16* Wkvt  = (_Float16*)(smem + 69632);
  _Float16* Os    = (_Float16*)(smem + 69632);
  _Float16* Woutt = (_Float16*)(smem + 104448);
  _Float16* Qs    = (_Float16*)(smem + 139264);
  _Float16* Ks    = (_Float16*)(smem + 174080);
  _Float16* Vts   = (_Float16*)(smem + 208896);

  int dir = blockIdx.y;
  const float* Wq   = dir ? Wq_w   : Wq_h;
  const float* Wkv  = dir ? Wkv_w  : Wkv_h;
  const float* Wout = dir ? Wout_w : Wout_h;
  const float* bout = dir ? bout_w : bout_h;
  float* out        = dir ? aw     : ah;

  int tid = threadIdx.x, lane = tid & 31, wid = tid >> 5;
  int b = blockIdx.x >> 7, s = blockIdx.x & 127;
  int rowstride = dir ? 128 : 16384;
  int base = b * 2097152 + s * (dir ? 16384 : 128);

  // -------- Phase A0: stage panel + Wq/Wkv (transposed, f16, padded strides) --------
  for (int i = tid; i < 128 * 128; i += 256) {
    int t = i >> 7, c = i & 127;
    Xs[t * LDH + c] = (_Float16)xc[base + t * rowstride + c];
  }
  for (int i = tid; i < 128 * 128; i += 256) {
    int k = i >> 7, n = i & 127;
    Wqt[n * LDH + k] = (_Float16)Wq[i];
  }
  for (int i = tid; i < 128 * 256; i += 256) {
    int k = i >> 8, n = i & 255;
    Wkvt[n * LDH + k] = (_Float16)Wkv[i];
  }
  __syncthreads();

  // -------- Phase A1: Q = X@Wq, [K|V] = X@Wkv (A fragments hoisted to registers) --------
  int m0 = wid * 16;
  {
    v16h xa[4];
#pragma unroll
    for (int i = 0; i < 4; ++i) xa[i] = ldA(Xs, LDH, m0, i * 32, lane);
#pragma unroll 1
    for (int nt = 0; nt < 8; ++nt) {  // Q
      v8f acc = {};
#pragma unroll
      for (int i = 0; i < 4; ++i)
        acc = wmma_f16(xa[i], ldBt(Wqt, LDH, i * 32, nt * 16, lane), acc);
      stD_f16(Qs, LDH, m0, nt * 16, acc, lane);
    }
#pragma unroll 1
    for (int nt = 0; nt < 8; ++nt) {  // K (cols 0..127 of Wkv)
      v8f acc = {};
#pragma unroll
      for (int i = 0; i < 4; ++i)
        acc = wmma_f16(xa[i], ldBt(Wkvt, LDH, i * 32, nt * 16, lane), acc);
      stD_f16(Ks, LDH, m0, nt * 16, acc, lane);
    }
#pragma unroll 1
    for (int nt = 0; nt < 8; ++nt) {  // V (cols 128..255), stored transposed
      v8f acc = {};
#pragma unroll
      for (int i = 0; i < 4; ++i)
        acc = wmma_f16(xa[i], ldBt(Wkvt, LDH, i * 32, 128 + nt * 16, lane), acc);
      stD_f16_T(Vts, LDH, m0, nt * 16, acc, lane);
    }
  }
  __syncthreads();

  // -------- Phase B: per head: dots -> softmax -> attn @ V --------
  for (int h = 0; h < 2; ++h) {
    const int co = h * 64;
    // dots = Q_h @ K_h^T * dh^-0.5  (K in LDS row-major == Bt for K^T)
    {
      v16h qa[2];
#pragma unroll
      for (int i = 0; i < 2; ++i) qa[i] = ldA(Qs, LDH, m0, co + i * 32, lane);
#pragma unroll 1
      for (int nt = 0; nt < 8; ++nt) {
        v8f acc = {};
#pragma unroll
        for (int i = 0; i < 2; ++i)
          acc = wmma_f16(qa[i], ldBt(Ks + co, LDH, i * 32, nt * 16, lane), acc);
        stD_f32s(DOTS, LDD, m0, nt * 16, acc, 0.125f, lane);
      }
    }
    // row softmax over own 16-row band (wave-private rows; shfl reductions, wave32)
#pragma unroll 1
    for (int r = 0; r < 16; ++r) {
      float* row = DOTS + (m0 + r) * LDD;
      float v0 = row[lane], v1 = row[lane + 32], v2 = row[lane + 64], v3 = row[lane + 96];
      float mx = fmaxf(fmaxf(v0, v1), fmaxf(v2, v3));
#pragma unroll
      for (int o = 16; o; o >>= 1) mx = fmaxf(mx, __shfl_xor(mx, o, 32));
      v0 = __expf(v0 - mx); v1 = __expf(v1 - mx); v2 = __expf(v2 - mx); v3 = __expf(v3 - mx);
      float sm = v0 + v1 + v2 + v3;
#pragma unroll
      for (int o = 16; o; o >>= 1) sm += __shfl_xor(sm, o, 32);
      float inv = 1.0f / sm;
      row[lane] = v0 * inv; row[lane + 32] = v1 * inv;
      row[lane + 64] = v2 * inv; row[lane + 96] = v3 * inv;
    }
    // out_h = attn @ V_h  (A from f32 LDS, B = Vts rows co..co+63)
    {
      v16h pa[4];
#pragma unroll
      for (int i = 0; i < 4; ++i) pa[i] = ldA_f32(DOTS, LDD, m0, i * 32, lane);
#pragma unroll 1
      for (int nt = 0; nt < 4; ++nt) {
        v8f acc = {};
#pragma unroll
        for (int i = 0; i < 4; ++i)
          acc = wmma_f16(pa[i], ldBt(Vts, LDH, i * 32, co + nt * 16, lane), acc);
        stD_f16(Os, LDH, m0, co + nt * 16, acc, lane);
      }
    }
    __syncthreads();
  }

  // -------- Phase C: A_out = O @ Wout + bout --------
  for (int i = tid; i < 128 * 128; i += 256) {
    int k = i >> 7, n = i & 127;
    Woutt[n * LDH + k] = (_Float16)Wout[i];
  }
  __syncthreads();
  {
    v16h oa[4];
#pragma unroll
    for (int i = 0; i < 4; ++i) oa[i] = ldA(Os, LDH, m0, i * 32, lane);
#pragma unroll 1
    for (int nt = 0; nt < 8; ++nt) {
      v8f acc = {};
#pragma unroll
      for (int i = 0; i < 4; ++i)
        acc = wmma_f16(oa[i], ldBt(Woutt, LDH, i * 32, nt * 16, lane), acc);
      int n = nt * 16 + (lane & 15);
      float bn = bout[n];
      int mb = m0 + ((lane & 16) ? 8 : 0);
#pragma unroll
      for (int r = 0; r < 8; ++r)
        out[base + (mb + r) * rowstride + n] = acc[r] + bn;
    }
  }
}

// ---------------- Kernel 3: y = relu( [relu(a_h+a_w) | xc] @ conv_w ) ----------------
__global__ __launch_bounds__(256) void conv_kernel(
    const float* __restrict__ ah, const float* __restrict__ aw,
    const float* __restrict__ xc, const float* __restrict__ conv_w,
    float* __restrict__ y)
{
  extern __shared__ char smem[];
  _Float16* As = (_Float16*)smem;               // 128 x LDC  (67584 B)
  _Float16* Wt = (_Float16*)(smem + 67584);     // 256 x LDC transposed (135168 B)
  int tid = threadIdx.x, lane = tid & 31, wid = tid >> 5;
  int p0 = blockIdx.x * 128;
  for (int i = tid; i < 128 * 128; i += 256) {
    int m = i >> 7, c = i & 127;
    float av = ah[(p0 + m) * 128 + c] + aw[(p0 + m) * 128 + c];
    As[m * LDC + c] = (_Float16)fmaxf(av, 0.0f);
    As[m * LDC + 128 + c] = (_Float16)xc[(p0 + m) * 128 + c];
  }
  for (int i = tid; i < 256 * 256; i += 256) {
    int k = i >> 8, n = i & 255;
    Wt[n * LDC + k] = (_Float16)conv_w[i];
  }
  __syncthreads();
  int m0 = wid * 16;
  {
    v16h aa[8];
#pragma unroll
    for (int i = 0; i < 8; ++i) aa[i] = ldA(As, LDC, m0, i * 32, lane);
#pragma unroll 1
    for (int nt = 0; nt < 16; ++nt) {
      v8f acc = {};
#pragma unroll
      for (int i = 0; i < 8; ++i)
        acc = wmma_f16(aa[i], ldBt(Wt, LDC, i * 32, nt * 16, lane), acc);
      int n = nt * 16 + (lane & 15);
      int mb = m0 + ((lane & 16) ? 8 : 0);
#pragma unroll
      for (int r = 0; r < 8; ++r)
        y[(p0 + mb + r) * 256 + n] = fmaxf(acc[r], 0.0f);
    }
  }
}

// ---------------- Kernel 4a: per-slice partial sums (deterministic, no atomics) ----------------
__global__ __launch_bounds__(256) void stats1_kernel(const float* __restrict__ y,
                                                     float* __restrict__ psum,
                                                     float* __restrict__ psq) {
  int e = threadIdx.x;
  int j = blockIdx.x;
  float s = 0.0f, q = 0.0f;
  const float* p = y + (size_t)j * 512 * 256 + e;
  for (int i = 0; i < 512; ++i) { float v = p[i * 256]; s += v; q += v * v; }
  psum[j * 256 + e] = s;
  psq[j * 256 + e]  = q;
}

// ---------------- Kernel 4b: finalize mean/var -> scale/shift ----------------
__global__ __launch_bounds__(256) void stats2_kernel(const float* __restrict__ psum,
                                                     const float* __restrict__ psq,
                                                     const float* __restrict__ gamma,
                                                     const float* __restrict__ beta,
                                                     float* __restrict__ scale,
                                                     float* __restrict__ shift) {
  int e = threadIdx.x;
  float s = 0.0f, q = 0.0f;
  for (int j = 0; j < 256; ++j) { s += psum[j * 256 + e]; q += psq[j * 256 + e]; }
  const float invN = 1.0f / 131072.0f;
  float mean = s * invN;
  float var = q * invN - mean * mean;
  float rstd = rsqrtf(var + 1e-5f);
  float g = gamma[e] * rstd;
  scale[e] = g;
  shift[e] = beta[e] - mean * g;
}

// ---------------- Kernel 5: BN apply + NHWC -> NCHW ----------------
__global__ __launch_bounds__(256) void apply_kernel(const float* __restrict__ y,
                                                    const float* __restrict__ scale,
                                                    const float* __restrict__ shift,
                                                    float* __restrict__ out) {
  __shared__ float tile[32][33];
  int e0 = blockIdx.x * 32;
  int w0 = blockIdx.y * 32;
  int bh = blockIdx.z;
  int b = bh >> 7, hh = bh & 127;
  int tid = threadIdx.x;
  for (int i = tid; i < 1024; i += 256) {
    int w = i >> 5, e = i & 31;
    tile[w][e] = y[((size_t)((b * 128 + hh) * 128 + w0 + w)) * 256 + e0 + e]
                 * scale[e0 + e] + shift[e0 + e];
  }
  __syncthreads();
  for (int i = tid; i < 1024; i += 256) {
    int e = i >> 5, w = i & 31;
    out[(((size_t)(b * 256 + e0 + e) * 128) + hh) * 128 + w0 + w] = tile[w][e];
  }
}

// ---------------- Host launcher ----------------
extern "C" void kernel_launch(void* const* d_in, const int* in_sizes, int n_in,
                              void* d_out, int out_size, void* d_ws, size_t ws_size,
                              hipStream_t stream) {
  (void)in_sizes; (void)n_in; (void)out_size; (void)ws_size;
  const float* x      = (const float*)d_in[0];
  const float* Wq_h   = (const float*)d_in[1];
  const float* Wkv_h  = (const float*)d_in[2];
  const float* Wout_h = (const float*)d_in[3];
  const float* bout_h = (const float*)d_in[4];
  const float* Wq_w   = (const float*)d_in[5];
  const float* Wkv_w  = (const float*)d_in[6];
  const float* Wout_w = (const float*)d_in[7];
  const float* bout_w = (const float*)d_in[8];
  const float* conv_w = (const float*)d_in[9];
  const float* gamma  = (const float*)d_in[10];
  const float* beta   = (const float*)d_in[11];
  float* out = (float*)d_out;

  // workspace layout (floats): xc(16.78M) ah(16.78M) aw(16.78M) y(33.55M) psum(64K) psq(64K) scale(256) shift(256)
  float* ws = (float*)d_ws;
  float* xc = ws;
  float* ah = xc + 16777216;
  float* aw = ah + 16777216;
  float* yb = aw + 16777216;
  float* ps = yb + 33554432;
  float* pq = ps + 65536;
  float* sc = pq + 65536;
  float* sh = sc + 256;

  // enable >64KB dynamic LDS (attribute set, not a stream op -> graph-capture safe)
  hipFuncSetAttribute((const void*)attn_kernel, hipFuncAttributeMaxDynamicSharedMemorySize, 243712);
  hipFuncSetAttribute((const void*)conv_kernel, hipFuncAttributeMaxDynamicSharedMemorySize, 202752);

  pool_kernel<<<dim3(4, 128, 32), 256, 0, stream>>>(x, xc);
  attn_kernel<<<dim3(1024, 2), 256, 243712, stream>>>(
      xc, Wq_h, Wkv_h, Wout_h, bout_h, Wq_w, Wkv_w, Wout_w, bout_w, ah, aw);
  conv_kernel<<<1024, 256, 202752, stream>>>(ah, aw, xc, conv_w, yb);
  stats1_kernel<<<256, 256, 0, stream>>>(yb, ps, pq);
  stats2_kernel<<<1, 256, 0, stream>>>(ps, pq, gamma, beta, sc, sh);
  apply_kernel<<<dim3(8, 4, 1024), 256, 0, stream>>>(yb, sc, sh, out);
}
